// RallyPredictor_12695923327319
// MI455X (gfx1250) — matrix operand order repaired and checked by reading, not compiled, not run on hardware
//
#include <hip/hip_runtime.h>
#include <hip/hip_bf16.h>

// ---------------------------------------------------------------------------
// Problem constants (reference: B=256, N=12, T=20, D=512)
// ---------------------------------------------------------------------------
#define BB 256
#define NN 12
#define TT 20
#define DD 512
#define EPS_A 1e-6f
#define EPS_BN 1e-5f
#define DIST2 (150.0f * 150.0f)

#define NT_TILES 32   // D / 16 output-column tiles
#define KT_TILES 16   // D / 32 k tiles

typedef __attribute__((ext_vector_type(16))) __bf16 v16bf;
typedef __attribute__((ext_vector_type(8)))  float  v8f;
typedef __attribute__((ext_vector_type(4)))  float  f4v;   // native vector for NT builtins

// ---------------------------------------------------------------------------
// Prep kernel: swizzle W1/W2 into WMMA B-matrix fragment order as bf16 hi/lo
// split pairs, and fold BN params into per-channel (scale, offset).
//
// B-fragment layout (32x16 bf16, wave32):  lane = (k/16)*16 + n_local,
// slot (0..15) = k % 16.  B[k][n] = W[d = nt*16+n][c = kt*32+k].
// Index: ((l*32 + nt)*16 + kt)*512 + lane*16 + slot
// ---------------------------------------------------------------------------
__global__ void prep_kernel(const float* __restrict__ W1,
                            const float* __restrict__ W2,
                            const float* __restrict__ b1, const float* __restrict__ g1,
                            const float* __restrict__ be1, const float* __restrict__ mm1,
                            const float* __restrict__ vv1,
                            const float* __restrict__ b2, const float* __restrict__ g2,
                            const float* __restrict__ be2, const float* __restrict__ mm2,
                            const float* __restrict__ vv2,
                            __bf16* __restrict__ whi, __bf16* __restrict__ wlo,
                            float* __restrict__ Sarr, float* __restrict__ Oarr)
{
    const int gid    = blockIdx.x * blockDim.x + threadIdx.x;
    const int stride = gridDim.x * blockDim.x;
    const int total  = 2 * DD * DD;              // 524288 swizzled elements
    for (int idx = gid; idx < total; idx += stride) {
        const int l    = idx >> 18;              // / 262144
        const int r    = idx & 262143;
        const int nt   = r >> 13;                // / (16*512)
        const int r2   = r & 8191;
        const int kt   = r2 >> 9;
        const int r3   = r2 & 511;
        const int lane = r3 >> 4;
        const int s    = r3 & 15;
        const int k    = ((lane >> 4) << 4) | s; // (lane/16)*16 + slot
        const int nl   = lane & 15;
        const int d    = nt * 16 + nl;
        const int c    = kt * 32 + k;
        const float* W = l ? W2 : W1;
        const float w  = W[d * DD + c];
        const __bf16 h = (__bf16)w;
        whi[idx] = h;                            // default TH: stays L2-resident
        wlo[idx] = (__bf16)(w - (float)h);
    }
    // Fold BN:  y = g*(acc + b - m)*rsqrt(v+eps) + beta = acc*S + O
    if (gid < 2 * DD) {
        const int l = gid >> 9, n = gid & 511;
        const float* g  = l ? g2  : g1;
        const float* b  = l ? b2  : b1;
        const float* be = l ? be2 : be1;
        const float* m  = l ? mm2 : mm1;
        const float* v  = l ? vv2 : vv1;
        const float sc = g[n] * rsqrtf(v[n] + EPS_BN);
        Sarr[gid] = sc;
        Oarr[gid] = (b[n] - m[n]) * sc + be[n];
    }
}

// ---------------------------------------------------------------------------
// Bulk copy: out = person_features (last time slice overwritten later).
// Pure HBM-bandwidth kernel. 252 MB of stream traffic > 192 MB L2, so use
// non-temporal cache policy (TH=NT) on both sides to avoid sweeping the L2
// that the STGCN kernel needs for its 2 MB of re-read weight fragments.
// ---------------------------------------------------------------------------
__global__ void copy_kernel(const f4v* __restrict__ src, f4v* __restrict__ dst, int n4)
{
    const int stride = gridDim.x * blockDim.x;
    for (int i = blockIdx.x * blockDim.x + threadIdx.x; i < n4; i += stride) {
        const f4v v = __builtin_nontemporal_load(&src[i]);
        __builtin_nontemporal_store(v, &dst[i]);
    }
}

// ---------------------------------------------------------------------------
// Main STGCN kernel: one block (256 thr = 8 wave32) per batch.
//   1) adjacency A_norm (12x12) from bbox centers
//   2) load x = PF[b,:,19,:] into LDS
//   3) aggregate xg = A_norm^T x, split to bf16 hi/lo, store pre-swizzled
//      A-fragments in LDS (16x32 bf16 layout per k-tile)
//   4) GEMM xg @ W^T via 3x v_wmma_f32_16x16x32_bf16 per k-step (bf16x2
//      split: AhiBhi + AhiBlo + AloBhi ~ fp32 accuracy), fused BN+relu
//      epilogue -> LDS (layer1) / global out (layer2)
// ---------------------------------------------------------------------------
__global__ __launch_bounds__(256) void stgcn_kernel(
    const float* __restrict__ pf, const float* __restrict__ bbox,
    const __bf16* __restrict__ whi, const __bf16* __restrict__ wlo,
    const float* __restrict__ Sarr, const float* __restrict__ Oarr,
    float* __restrict__ out)
{
    __shared__ float sA[NN * 16];                        // A_norm[n][m]
    __shared__ float sInv[NN];
    __shared__ float sX[NN * DD];                        // activations, 24KB
    __shared__ __attribute__((aligned(32))) __bf16 sFragHi[KT_TILES * 512];  // 16KB
    __shared__ __attribute__((aligned(32))) __bf16 sFragLo[KT_TILES * 512];  // 16KB

    const int b    = blockIdx.x;
    const int t    = threadIdx.x;
    const int lane = t & 31;
    const int wave = t >> 5;

    // --- adjacency ---------------------------------------------------------
    if (t < NN * NN) {
        const int i = t / NN, j = t % NN;
        const float* bi = bbox + (b * NN + i) * 4;
        const float* bj = bbox + (b * NN + j) * 4;
        const float cxi = bi[0] + 0.5f * bi[2], cyi = bi[1] + 0.5f * bi[3];
        const float cxj = bj[0] + 0.5f * bj[2], cyj = bj[1] + 0.5f * bj[3];
        const float dx = cxi - cxj, dy = cyi - cyj;
        sA[i * 16 + j] = (dx * dx + dy * dy < DIST2) ? 1.0f : 0.0f;
    }
    // --- load last-slice activations --------------------------------------
    for (int e = t; e < NN * (DD / 4); e += 256) {       // 1536 float4
        const int n = e >> 7, q = e & 127;
        const f4v v = *(const f4v*)(pf + ((size_t)((b * NN + n) * TT + TT - 1)) * DD + q * 4);
        *(f4v*)(sX + n * DD + q * 4) = v;
    }
    __syncthreads();
    if (t < NN) {
        float s = EPS_A;
        #pragma unroll
        for (int j = 0; j < NN; ++j) s += sA[t * 16 + j];
        sInv[t] = 1.0f / s;
    }
    __syncthreads();
    if (t < NN * NN) { const int i = t / NN, j = t % NN; sA[i * 16 + j] *= sInv[i]; }
    __syncthreads();

    for (int layer = 0; layer < 2; ++layer) {
        // --- aggregate + split + swizzle into A fragments ------------------
        // A frag (16x32 bf16): elem (m, kk): slot = kk&15,
        // lane = m + 16*(kk>>4)
        for (int e = t; e < 16 * DD; e += 256) {
            const int m = e >> 9, k = e & 511;
            float val = 0.0f;
            if (m < NN) {
                float acc = 0.0f;
                #pragma unroll
                for (int n = 0; n < NN; ++n) acc += sA[n * 16 + m] * sX[n * DD + k];
                val = acc;
            }
            const __bf16 h = (__bf16)val;
            const __bf16 lo = (__bf16)(val - (float)h);
            const int kt = k >> 5, kk = k & 31;
            const int s  = kk & 15;
            const int ln = m + ((kk >> 4) << 4);
            const int fi = kt * 512 + ln * 16 + s;
            sFragHi[fi] = h;
            sFragLo[fi] = lo;
        }
        __syncthreads();

        // --- GEMM: 4 column-tiles per wave, 16 k-tiles, 3 WMMAs each -------
        const __bf16* wHiL = whi + (size_t)layer * (NT_TILES * KT_TILES * 512);
        const __bf16* wLoL = wlo + (size_t)layer * (NT_TILES * KT_TILES * 512);
        const float*  Sl   = Sarr + layer * DD;
        const float*  Ol   = Oarr + layer * DD;

        for (int ti = 0; ti < 4; ++ti) {
            const int ct = wave * 4 + ti;                // column tile 0..31
            v8f acc = {};
            const __bf16* wh = wHiL + (size_t)ct * (KT_TILES * 512);
            const __bf16* wl = wLoL + (size_t)ct * (KT_TILES * 512);
            for (int kt = 0; kt < KT_TILES; ++kt) {
                const v16bf a_hi = *(const v16bf*)&sFragHi[kt * 512 + lane * 16];
                const v16bf a_lo = *(const v16bf*)&sFragLo[kt * 512 + lane * 16];
                const v16bf b_hi = *(const v16bf*)&wh[kt * 512 + lane * 16];
                const v16bf b_lo = *(const v16bf*)&wl[kt * 512 + lane * 16];
                acc = __builtin_amdgcn_wmma_f32_16x16x32_bf16(false, a_hi, false, b_hi,
                                                              (short)0, acc, false, false);
                acc = __builtin_amdgcn_wmma_f32_16x16x32_bf16(false, a_hi, false, b_lo,
                                                              (short)0, acc, false, false);
                acc = __builtin_amdgcn_wmma_f32_16x16x32_bf16(false, a_lo, false, b_hi,
                                                              (short)0, acc, false, false);
            }
            // epilogue: BN + relu.  C/D layout: VGPR r, lanes 0-15 -> M=r,
            // lanes 16-31 -> M=8+r; N = ct*16 + (lane&15).
            const int n  = ct * 16 + (lane & 15);
            const float Sv = Sl[n], Ov = Ol[n];
            const int mbase = (lane >> 4) * 8;
            #pragma unroll
            for (int r = 0; r < 8; ++r) {
                const int m = mbase + r;
                if (m < NN) {
                    const float y = fmaxf(acc[r] * Sv + Ov, 0.0f);
                    if (layer == 0)
                        sX[m * DD + n] = y;
                    else
                        out[((size_t)((b * NN + m) * TT + TT - 1)) * DD + n] = y;
                }
            }
        }
        __syncthreads();
    }
}

// ---------------------------------------------------------------------------
// Host-side launch.  Input order (setup_inputs dict):
// 0 person_features, 1 bboxes, 2 W1, 3 b1, 4 g1, 5 beta1, 6 m1, 7 v1,
// 8 W2, 9 b2, 10 g2, 11 beta2, 12 m2, 13 v2
// Workspace: [0,1MB) whi, [1MB,2MB) wlo, [2MB,+4KB) S, [+4KB,+8KB) O
// ---------------------------------------------------------------------------
extern "C" void kernel_launch(void* const* d_in, const int* in_sizes, int n_in,
                              void* d_out, int out_size, void* d_ws, size_t ws_size,
                              hipStream_t stream)
{
    const float* pf   = (const float*)d_in[0];
    const float* bbox = (const float*)d_in[1];
    const float* W1 = (const float*)d_in[2],  *b1 = (const float*)d_in[3];
    const float* g1 = (const float*)d_in[4],  *be1 = (const float*)d_in[5];
    const float* m1 = (const float*)d_in[6],  *v1 = (const float*)d_in[7];
    const float* W2 = (const float*)d_in[8],  *b2 = (const float*)d_in[9];
    const float* g2 = (const float*)d_in[10], *be2 = (const float*)d_in[11];
    const float* m2 = (const float*)d_in[12], *v2 = (const float*)d_in[13];

    char* ws = (char*)d_ws;
    __bf16* whi = (__bf16*)ws;
    __bf16* wlo = (__bf16*)(ws + (1u << 20));
    float*  Sarr = (float*)(ws + (2u << 20));
    float*  Oarr = (float*)(ws + (2u << 20) + 4096);

    prep_kernel<<<512, 256, 0, stream>>>(W1, W2, b1, g1, be1, m1, v1,
                                         b2, g2, be2, m2, v2, whi, wlo, Sarr, Oarr);

    const int n4 = BB * NN * TT * DD / 4;   // 7,864,320 float4
    copy_kernel<<<4096, 256, 0, stream>>>((const f4v*)pf, (f4v*)d_out, n4);

    stgcn_kernel<<<BB, 256, 0, stream>>>(pf, bbox, whi, wlo, Sarr, Oarr, (float*)d_out);
}
